// fMRI3DGNN_86784109183047
// MI455X (gfx1250) — matrix operand order repaired and checked by reading, not compiled, not run on hardware
//
#include <hip/hip_runtime.h>
#include <math.h>

// ---------------------------------------------------------------------------
// Types for CDNA5 WMMA (wave32)
// ---------------------------------------------------------------------------
typedef __attribute__((ext_vector_type(16))) __bf16 v16bf;
typedef __attribute__((ext_vector_type(8)))  float  v8f;

#define NN   116
#define BB   8
#define NN2  13456   // 116*116
#define KGB  6670

__device__ __forceinline__ float gelu_f(float x) {
    return 0.5f * x * (1.0f + erff(x * 0.70710678118654752f));
}

// ---------------------------------------------------------------------------
// bf16 WMMA GEMM, templated on K and N so the hot loop has:
//   - compile-time trip count, no per-element bounds branches
//   - W column loads as one base pointer + immediate offsets (t*N folds)
// C[M,N] = A[M,K] @ W[K,N] + bias
//   mode 0: plain (bias add)
//   mode 1: sigmoid + threshold at 0.5 (adjacency builder)
// One wave per 16x16 C tile; 4 waves (128 thr) per block cover 64 columns.
// Rows m >= M are computed on clamped (row 0) data and simply never stored.
// ---------------------------------------------------------------------------
template <int K, int N>
__global__ __launch_bounds__(128) void gemm_bf16_wmma(
    const float* __restrict__ A, const float* __restrict__ W,
    const float* __restrict__ bias, float* __restrict__ C,
    int M, int mode)
{
    const int wave = threadIdx.x >> 5;
    const int lane = threadIdx.x & 31;
    const int half = lane >> 4;
    const int r    = lane & 15;

    const int n0 = blockIdx.x * 64 + wave * 16;
    const int m0 = blockIdx.y * 16;
    if (n0 >= N) return;

    const int m = m0 + r;                 // A row this lane stages
    const int n = n0 + r;                 // W column this lane stages
    const float* Arow = A + (size_t)(m < M ? m : 0) * (size_t)K;

    v8f acc = {0.f, 0.f, 0.f, 0.f, 0.f, 0.f, 0.f, 0.f};

    constexpr int KMAIN = K & ~31;

    for (int kb = 0; kb < KMAIN; kb += 32) {
        v16bf av, bv;
        // A operand: contiguous k order, same as B (half selects K 0..15 vs 16..31)
        const float* Ap = Arow + kb + half * 16;
#pragma unroll
        for (int t = 0; t < 8; ++t) {
            av[t]     = (__bf16)Ap[t];
            av[t + 8] = (__bf16)Ap[8 + t];
        }
        // B operand: contiguous k order, same as A (half selects K 0..15 vs 16..31)
        const float* Wp = W + (size_t)(kb + half * 16) * (size_t)N + n;
#pragma unroll
        for (int t = 0; t < 16; ++t)
            bv[t] = (__bf16)Wp[(size_t)t * N];     // immediate offsets (N const)

        if (kb + 32 < K)
            __builtin_prefetch(Wp + (size_t)32 * N, 0, 1);

        acc = __builtin_amdgcn_wmma_f32_16x16x32_bf16(
                  false, av, false, bv, (short)0, acc, false, false);
    }

    if constexpr ((K & 31) != 0) {
        // Branch-free tail: unconditional loads on clamped indices + selects.
        constexpr int kb = KMAIN;
        v16bf av, bv;
        const int ka = kb + half * 16;
#pragma unroll
        for (int t = 0; t < 8; ++t) {
            int k0 = ka + t;
            int k1 = ka + 8 + t;
            float a0 = Arow[k0 < K ? k0 : 0];
            float a1 = Arow[k1 < K ? k1 : 0];
            av[t]     = (__bf16)(k0 < K ? a0 : 0.f);
            av[t + 8] = (__bf16)(k1 < K ? a1 : 0.f);
        }
        const int kw = kb + half * 16;
#pragma unroll
        for (int t = 0; t < 16; ++t) {
            int k0 = kw + t;
            float w0 = W[(size_t)(k0 < K ? k0 : 0) * (size_t)N + n];
            bv[t] = (__bf16)(k0 < K ? w0 : 0.f);
        }
        acc = __builtin_amdgcn_wmma_f32_16x16x32_bf16(
                  false, av, false, bv, (short)0, acc, false, false);
    }

    // Epilogue: VGPR rr holds row m0 + rr + 8*half, column n0 + r
    const float bval = bias ? bias[n] : 0.f;
#pragma unroll
    for (int rr = 0; rr < 8; ++rr) {
        int mo = m0 + rr + 8 * half;
        if (mo < M) {
            float v = acc[rr] + bval;
            if (mode == 1) {
                float s = 1.f / (1.f + expf(-v));
                v = (s > 0.5f) ? s : 0.f;   // thresholded adjacency
            }
            C[(size_t)mo * (size_t)N + n] = v;
        }
    }
}

// ---------------------------------------------------------------------------
// Per-row stats of thresholded adjacency: mu (mean over 116) and unbiased std
// One wave per (b, i) row.
// ---------------------------------------------------------------------------
__global__ __launch_bounds__(32) void stats_kernel(
    const float* __restrict__ adj_t, float* __restrict__ stats)
{
    const int row  = blockIdx.x;                // b*116 + i
    const int lane = threadIdx.x;
    const float* p = adj_t + (size_t)(row / NN) * NN2 + (size_t)(row % NN) * NN;

    float v[4];
    float s = 0.f;
#pragma unroll
    for (int t = 0; t < 4; ++t) {
        int j = lane + 32 * t;
        v[t] = (j < NN) ? p[j] : 0.f;
        s += v[t];
    }
#pragma unroll
    for (int off = 16; off > 0; off >>= 1) s += __shfl_xor(s, off, 32);
    float mu = s / (float)NN;

    float q = 0.f;
#pragma unroll
    for (int t = 0; t < 4; ++t) {
        int j = lane + 32 * t;
        float d = v[t] - mu;
        if (j < NN) q += d * d;
    }
#pragma unroll
    for (int off = 16; off > 0; off >>= 1) q += __shfl_xor(q, off, 32);

    if (lane == 0) {
        stats[row * 2 + 0] = mu;
        stats[row * 2 + 1] = sqrtf(q / (float)(NN - 1));
    }
}

// ---------------------------------------------------------------------------
// Feature extractor MLP (2 -> 8 gelu -> 16) + LayerNorm(16). One thread/row.
// ---------------------------------------------------------------------------
__global__ void feat_kernel(
    const float* __restrict__ stats,
    const float* __restrict__ fe_W1, const float* __restrict__ fe_b1,
    const float* __restrict__ fe_W2, const float* __restrict__ fe_b2,
    const float* __restrict__ ln_g,  const float* __restrict__ ln_b,
    float* __restrict__ h0)
{
    const int row = blockIdx.x * blockDim.x + threadIdx.x;
    if (row >= BB * NN) return;

    const float mu = stats[row * 2 + 0];
    const float sd = stats[row * 2 + 1];

    float hk[8];
#pragma unroll
    for (int k = 0; k < 8; ++k) {
        float z = mu * fe_W1[k] + sd * fe_W1[8 + k] + fe_b1[k];
        hk[k] = gelu_f(z);
    }
    float o[16];
    float lm = 0.f;
#pragma unroll
    for (int f = 0; f < 16; ++f) {
        float a = fe_b2[f];
#pragma unroll
        for (int k = 0; k < 8; ++k) a += hk[k] * fe_W2[k * 16 + f];
        o[f] = a;
        lm += a;
    }
    lm *= (1.f / 16.f);
    float lv = 0.f;
#pragma unroll
    for (int f = 0; f < 16; ++f) { float d = o[f] - lm; lv += d * d; }
    lv *= (1.f / 16.f);
    const float inv = rsqrtf(lv + 1e-5f);
#pragma unroll
    for (int f = 0; f < 16; ++f)
        h0[(size_t)row * 16 + f] = (o[f] - lm) * inv * ln_g[f] + ln_b[f];
}

// ---------------------------------------------------------------------------
// GATv2 attention (LeakyReLU inside the score -> not GEMM-decomposable).
// One block per (target i, head, batch); blockDim.x == d.
// Fused: masked softmax over sources + aggregation + bias + GELU.
// ---------------------------------------------------------------------------
__global__ __launch_bounds__(512) void gatv2_attn(
    const float* __restrict__ xl, const float* __restrict__ xr,
    const float* __restrict__ att, const float* __restrict__ bias,
    const float* __restrict__ adj_t, float* __restrict__ hout,
    int heads, int d, int self_loops)
{
    const int i    = blockIdx.x;
    const int head = blockIdx.y;
    const int b    = blockIdx.z;
    const int HD   = heads * d;
    const int tid  = threadIdx.x;

    __shared__ float         sh_e[128];
    __shared__ unsigned char sh_m[128];
    __shared__ float         sh_red;

    const size_t rowi = ((size_t)b * NN + i) * (size_t)HD + (size_t)head * d;

    if (tid < 128) { sh_e[tid] = -1e30f; sh_m[tid] = 0; }
    __syncthreads();

    if (tid < NN) {
        const int j = tid;
        bool m = (adj_t[(size_t)b * NN2 + (size_t)j * NN + i] > 0.5f);
        if (self_loops && j == i) m = true;
        sh_m[j] = m ? 1 : 0;
        if (m) {
            const size_t rowj = ((size_t)b * NN + j) * (size_t)HD + (size_t)head * d;
            float acc = 0.f;
            for (int o = 0; o < d; ++o) {
                float z = xr[rowi + o] + xl[rowj + o];
                z = (z > 0.f) ? z : 0.2f * z;         // leaky_relu 0.2
                acc += att[head * d + o] * z;
            }
            sh_e[j] = acc;
        }
    }
    __syncthreads();

    if (tid == 0) {
        float mx = -1e30f;
        for (int j = 0; j < NN; ++j) mx = fmaxf(mx, sh_e[j]);
        sh_red = mx;
    }
    __syncthreads();
    if (tid < NN) {
        float p = sh_m[tid] ? expf(sh_e[tid] - sh_red) : 0.f;
        sh_e[tid] = p;
    }
    __syncthreads();
    if (tid == 0) {
        float sm = 0.f;
        for (int j = 0; j < NN; ++j) sm += sh_e[j];
        sh_red = 1.f / fmaxf(sm, 1e-16f);
    }
    __syncthreads();

    // aggregation: out[., tid] = sum_j alpha_j * xl[j, tid]
    {
        const float inv = sh_red;
        float acc = 0.f;
        for (int j = 0; j < NN; ++j) {
            float a = sh_e[j] * inv;
            acc += a * xl[((size_t)b * NN + j) * (size_t)HD + (size_t)head * d + tid];
        }
        float v = acc + bias[head * d + tid];
        hout[rowi + tid] = gelu_f(v);
    }
}

// ---------------------------------------------------------------------------
// Mean-pool over nodes + MLP head (512 -> 256 BN/GELU -> 2). One block/batch.
// ---------------------------------------------------------------------------
__global__ __launch_bounds__(256) void classifier_kernel(
    const float* __restrict__ h,          // [B*116, 512]
    const float* __restrict__ Wc1, const float* __restrict__ bc1,
    const float* __restrict__ bn_g, const float* __restrict__ bn_b,
    const float* __restrict__ Wc2, const float* __restrict__ bc2,
    float* __restrict__ out)
{
    const int b = blockIdx.x;
    const int t = threadIdx.x;

    __shared__ float pooled[512];
    __shared__ float c1[256];

    for (int c = t; c < 512; c += 256) {
        float s = 0.f;
        for (int i = 0; i < NN; ++i)
            s += h[((size_t)b * NN + i) * 512 + c];
        pooled[c] = s * (1.f / (float)NN);
    }
    __syncthreads();

    {
        float a = bc1[t];
        for (int k = 0; k < 512; ++k) a += pooled[k] * Wc1[k * 256 + t];
        a = a * (1.f / sqrtf(1.f + 1e-5f)) * bn_g[t] + bn_b[t];
        c1[t] = gelu_f(a);
    }
    __syncthreads();

    if (t < 2) {
        float a = bc2[t];
        for (int k = 0; k < 256; ++k) a += c1[k] * Wc2[k * 2 + t];
        out[b * 2 + t] = a;
    }
}

// ---------------------------------------------------------------------------
// Host-side orchestration
// ---------------------------------------------------------------------------
extern "C" void kernel_launch(void* const* d_in, const int* in_sizes, int n_in,
                              void* d_out, int out_size, void* d_ws, size_t ws_size,
                              hipStream_t stream)
{
    (void)in_sizes; (void)n_in; (void)out_size; (void)ws_size;

    const float* x     = (const float*)d_in[0];
    const float* W_gb  = (const float*)d_in[1];
    const float* b_gb  = (const float*)d_in[2];
    const float* fe_W1 = (const float*)d_in[3];
    const float* fe_b1 = (const float*)d_in[4];
    const float* fe_W2 = (const float*)d_in[5];
    const float* fe_b2 = (const float*)d_in[6];
    const float* ln_g  = (const float*)d_in[7];
    const float* ln_b  = (const float*)d_in[8];
    const float* Wl[3]   = {(const float*)d_in[9],  (const float*)d_in[15], (const float*)d_in[21]};
    const float* bl[3]   = {(const float*)d_in[10], (const float*)d_in[16], (const float*)d_in[22]};
    const float* Wr[3]   = {(const float*)d_in[11], (const float*)d_in[17], (const float*)d_in[23]};
    const float* br[3]   = {(const float*)d_in[12], (const float*)d_in[18], (const float*)d_in[24]};
    const float* att[3]  = {(const float*)d_in[13], (const float*)d_in[19], (const float*)d_in[25]};
    const float* bias[3] = {(const float*)d_in[14], (const float*)d_in[20], (const float*)d_in[26]};
    const float* Wc1  = (const float*)d_in[27];
    const float* bc1  = (const float*)d_in[28];
    const float* bn_g = (const float*)d_in[29];
    const float* bn_b = (const float*)d_in[30];
    const float* Wc2  = (const float*)d_in[31];
    const float* bc2  = (const float*)d_in[32];
    float* out = (float*)d_out;

    // workspace layout (floats)
    float* ws = (float*)d_ws;
    size_t off = 0;
    float* adj   = ws + off; off += (size_t)BB * NN2;        // 107648
    float* stats = ws + off; off += (size_t)BB * NN * 2;     // 1856
    float* h0    = ws + off; off += (size_t)BB * NN * 16;    // 14848
    float* xlb   = ws + off; off += (size_t)BB * NN * 1024;  // 950272
    float* xrb   = ws + off; off += (size_t)BB * NN * 1024;
    float* hA    = ws + off; off += (size_t)BB * NN * 1024;
    float* hB    = ws + off; off += (size_t)BB * NN * 1024;

    // 1) adjacency builder: sigmoid(x @ W_gb + b) thresholded   (HBM-bound)
    {
        dim3 grid((NN2 + 63) / 64, (BB + 15) / 16);
        gemm_bf16_wmma<KGB, NN2><<<grid, 128, 0, stream>>>(x, W_gb, b_gb, adj, BB, 1);
    }
    // 2) per-row mean / unbiased std
    stats_kernel<<<BB * NN, 32, 0, stream>>>(adj, stats);
    // 3) feature MLP + LayerNorm -> h0 [928,16]
    feat_kernel<<<(BB * NN + 127) / 128, 128, 0, stream>>>(
        stats, fe_W1, fe_b1, fe_W2, fe_b2, ln_g, ln_b, h0);

    const int M = BB * NN;                // 928
    dim3 gM((1024 + 63) / 64, (M + 15) / 16);
    dim3 gM512((512 + 63) / 64, (M + 15) / 16);

    // 4) GATv2 layer 0:  h0[928,16] -> xl/xr[928,1024], heads=8, d=128
    gemm_bf16_wmma<16, 1024><<<gM, 128, 0, stream>>>(h0, Wl[0], bl[0], xlb, M, 0);
    gemm_bf16_wmma<16, 1024><<<gM, 128, 0, stream>>>(h0, Wr[0], br[0], xrb, M, 0);
    {
        dim3 agrid(NN, 8, BB);
        gatv2_attn<<<agrid, 128, 0, stream>>>(xlb, xrb, att[0], bias[0], adj, hA, 8, 128, 0);
    }
    // 5) GATv2 layer 1:  hA[928,1024] -> xl/xr[928,1024], heads=4, d=256
    gemm_bf16_wmma<1024, 1024><<<gM, 128, 0, stream>>>(hA, Wl[1], bl[1], xlb, M, 0);
    gemm_bf16_wmma<1024, 1024><<<gM, 128, 0, stream>>>(hA, Wr[1], br[1], xrb, M, 0);
    {
        dim3 agrid(NN, 4, BB);
        gatv2_attn<<<agrid, 256, 0, stream>>>(xlb, xrb, att[1], bias[1], adj, hB, 4, 256, 1);
    }
    // 6) GATv2 layer 2:  hB[928,1024] -> xl/xr[928,512], heads=1, d=512
    gemm_bf16_wmma<1024, 512><<<gM512, 128, 0, stream>>>(hB, Wl[2], bl[2], xlb, M, 0);
    gemm_bf16_wmma<1024, 512><<<gM512, 128, 0, stream>>>(hB, Wr[2], br[2], xrb, M, 0);
    {
        dim3 agrid(NN, 1, BB);
        gatv2_attn<<<agrid, 512, 0, stream>>>(xlb, xrb, att[2], bias[2], adj, hA, 1, 512, 1);
    }

    // 7) mean-pool + classifier head -> [8,2]
    classifier_kernel<<<BB, 256, 0, stream>>>(hA, Wc1, bc1, bn_g, bn_b, Wc2, bc2, out);
}
